// AttentionFlow_420906795220
// MI455X (gfx1250) — matrix operand order repaired and checked by reading, not compiled, not run on hardware
//
#include <hip/hip_runtime.h>
#include <hip/hip_bf16.h>
#include <stdint.h>

#define E_EDGES 131072
#define BQ 64
#define N_MEMSZ 100000
#define N_SEGS 50000
#define DD 512
#define DSMALL 128

typedef __attribute__((ext_vector_type(16))) __bf16 v16bf;
typedef __attribute__((ext_vector_type(8)))  float  v8f;

union Frag {
  v16bf v;
  unsigned int u[8];
  unsigned short s[16];
};

// cheap fp32 -> bf16 (round-half-up; bias << bf16 quantization error)
__device__ __forceinline__ unsigned short f32_bf16(float f) {
  return (unsigned short)((__float_as_uint(f) + 0x8000u) >> 16);
}

// pack two fp32 into a bf16 pair with one v_perm_b32
__device__ __forceinline__ unsigned int pack2_bf16(float lo, float hi) {
  unsigned int a = __float_as_uint(lo) + 0x8000u;
  unsigned int b = __float_as_uint(hi) + 0x8000u;
  return __builtin_amdgcn_perm(b, a, 0x07060302u);  // {b[31:16], a[31:16]}
}

// A-fragment (16x32 bf16): lanes 0-15 = K{0..15},
// lanes 16-31 = K{16..31}; `row` points at element kbase of lane's M-row.
__device__ __forceinline__ void loadA_bf16(Frag& a, const unsigned short* row, int lane) {
  const int half = (lane >> 4) & 1;
#pragma unroll
  for (int v = 0; v < 8; ++v) {
    const int k = half * 16 + 2 * v;
    a.u[v] = *(const unsigned int*)(row + k);
  }
}

// B-fragment (32x16 bf16): B[k][n] = W[n][k], W row-major [N x Kdim] bf16.
__device__ __forceinline__ void loadB(Frag& b, const unsigned short* W, int Kdim,
                                      int nbase, int kbase, int lane) {
  const int n = nbase + (lane & 15);
  const int koff = ((lane >> 4) & 1) * 16;
  const unsigned short* p = W + (size_t)n * Kdim + kbase + koff;
#pragma unroll
  for (int v = 0; v < 8; ++v) b.u[v] = *(const unsigned int*)(p + 2 * v);
}

__device__ __forceinline__ v8f wmma_bf16(const Frag& a, const Frag& b, v8f c) {
  return __builtin_amdgcn_wmma_f32_16x16x32_bf16(false, a.v, false, b.v,
                                                 (short)0, c, false, false);
}

__device__ __forceinline__ float leaky(float x) { return x >= 0.f ? x : 0.01f * x; }

// order-preserving float <-> uint map for atomicMax-based segment max
__device__ __forceinline__ unsigned int fmap(float f) {
  unsigned int u = __float_as_uint(f);
  return (u & 0x80000000u) ? ~u : (u | 0x80000000u);
}
__device__ __forceinline__ float funmap(unsigned int u) {
  unsigned int v = (u & 0x80000000u) ? (u & 0x7FFFFFFFu) : ~u;
  return __uint_as_float(v);
}

// ---------------------------------------------------------------- utilities
__global__ void k_cvt_bf16(const float* __restrict__ src,
                           unsigned short* __restrict__ dst, int n) {
  int i = blockIdx.x * blockDim.x + threadIdx.x;
  if (i < n) dst[i] = f32_bf16(src[i]);
}

__global__ void k_zero_f32(float* __restrict__ p, int n) {
  int i = blockIdx.x * blockDim.x + threadIdx.x;
  if (i < n) p[i] = 0.f;
}

__global__ void k_init_seg(unsigned int* __restrict__ segmax,
                           float* __restrict__ denom) {
  int i = blockIdx.x * blockDim.x + threadIdx.x;
  if (i < N_SEGS) {
    segmax[i] = 0x007FFFFFu;   // fmap(-inf)
    denom[i] = 0.f;
  }
}

// tiny: q_small[t][b][j] = dot(query_t[b,:], Wproj[j,:]) + bproj[j]
__global__ void k_qproj(const float* __restrict__ qsrc, const float* __restrict__ qrel,
                        const float* __restrict__ qtime, const float* __restrict__ Wproj,
                        const float* __restrict__ bproj, float* __restrict__ q_small) {
  int i = blockIdx.x * blockDim.x + threadIdx.x;
  if (i >= 3 * BQ * DSMALL) return;
  int t = i / (BQ * DSMALL);
  int b = (i / DSMALL) % BQ;
  int j = i % DSMALL;
  const float* q = (t == 0 ? qsrc : (t == 1 ? qrel : qtime)) + (size_t)b * DD;
  const float* w = Wproj + (size_t)j * DD;
  float acc = 0.f;
  for (int k = 0; k < DD; k += 4)
    acc += q[k] * w[k] + q[k + 1] * w[k + 1] + q[k + 2] * w[k + 2] + q[k + 3] * w[k + 3];
  q_small[i] = acc + bproj[j];
}

// -------------------------------------------------- fused per-edge logits
// one block = 16 edges, 4 waves; all GEMMs via v_wmma_f32_16x16x32_bf16
__global__ __launch_bounds__(128) void k_logits(
    const float* __restrict__ memorized, const float* __restrict__ rel_emb,
    const int* __restrict__ eg_idx, const int* __restrict__ vi_idx,
    const int* __restrict__ vj_idx,
    const unsigned short* __restrict__ Wproj_bf, const float* __restrict__ bproj,
    const unsigned short* __restrict__ Wl_bf, const float* __restrict__ bl,
    const unsigned short* __restrict__ Wr_bf, const float* __restrict__ br,
    const unsigned short* __restrict__ Wc_bf, const float* __restrict__ bc,
    const float* __restrict__ q_small, float* __restrict__ logits) {
  // phase A reuses the 48 KB stage as phase-B/C scratch (barrier-separated)
  __shared__ union {
    unsigned short stage[3][16][DD];           // 48 KB: hvi/hvj/rel rows (bf16)
    struct {
      unsigned short small[16][6 * DSMALL];    // 24 KB: hvi_s,hvj_s,rel_s,qs,qr,qt
      float lx[16][DSMALL];                    // 8 KB
      unsigned short rx1[16][DSMALL];          // 4 KB
    } p;
  } sh;
  __shared__ float lds_logit[16];

  const int e0 = blockIdx.x * 16;
  const int tid = threadIdx.x;
  const int lane = tid & 31;
  const int wave = tid >> 5;                 // 0..3, each owns 2 N-tiles
  const int m16 = lane & 15;
  const int m0 = ((lane >> 4) & 1) * 8;
  const int nlo = lane & 15;
  const v8f zero8 = {0.f, 0.f, 0.f, 0.f, 0.f, 0.f, 0.f, 0.f};

  // ---- stage gathered rows as bf16 in LDS (convert ONCE per block) ----
  for (int i = tid; i < 3 * 16 * (DD / 2); i += 128) {
    const int t = i / (16 * (DD / 2));
    const int rem = i % (16 * (DD / 2));
    const int e = rem / (DD / 2);
    const int kp = rem % (DD / 2);           // bf16 pair index
    const float* src;
    if (t == 0)      src = memorized + (size_t)vi_idx[e0 + e] * DD;
    else if (t == 1) src = memorized + (size_t)vj_idx[e0 + e] * DD;
    else             src = rel_emb + (size_t)(e0 + e) * DD;
    *(unsigned int*)&sh.stage[t][e][2 * kp] = pack2_bf16(src[2 * kp], src[2 * kp + 1]);
  }
  __syncthreads();

  // ---- phase A: hvi_s / hvj_s / rel_s = proj(...)  (K = 512) ----
  v8f aHvi[2], aHvj[2], aRel[2];
#pragma unroll
  for (int nt = 0; nt < 2; ++nt) { aHvi[nt] = zero8; aHvj[nt] = zero8; aRel[nt] = zero8; }

  for (int ks = 0; ks < 16; ++ks) {
    const int kbase = ks * 32;
    Frag fHvi, fHvj, fRel;
    loadA_bf16(fHvi, &sh.stage[0][m16][kbase], lane);
    loadA_bf16(fHvj, &sh.stage[1][m16][kbase], lane);
    loadA_bf16(fRel, &sh.stage[2][m16][kbase], lane);
#pragma unroll
    for (int nt = 0; nt < 2; ++nt) {
      Frag fB;
      loadB(fB, Wproj_bf, DD, (wave * 2 + nt) * 16, kbase, lane);
      aHvi[nt] = wmma_bf16(fHvi, fB, aHvi[nt]);
      aHvj[nt] = wmma_bf16(fHvj, fB, aHvj[nt]);
      aRel[nt] = wmma_bf16(fRel, fB, aRel[nt]);
    }
  }
  __syncthreads();   // all reads of sh.stage done; safe to overwrite as sh.p

  // epilogue A: write small matrices; gather qs/qr/qt; zero logit accum
#pragma unroll
  for (int nt = 0; nt < 2; ++nt) {
    const int n = (wave * 2 + nt) * 16 + nlo;
    const float bp = bproj[n];
#pragma unroll
    for (int r = 0; r < 8; ++r) {
      const int m = m0 + r;
      sh.p.small[m][0 * DSMALL + n] = f32_bf16(aHvi[nt][r] + bp);
      sh.p.small[m][1 * DSMALL + n] = f32_bf16(aHvj[nt][r] + bp);
      sh.p.small[m][2 * DSMALL + n] = f32_bf16(aRel[nt][r] + bp);
    }
  }
  for (int i = tid; i < 16 * 3 * DSMALL; i += 128) {
    const int e = i / (3 * DSMALL);
    const int r = i % (3 * DSMALL);
    const int t = r / DSMALL;
    const int n = r % DSMALL;
    const int b = eg_idx[e0 + e];
    sh.p.small[e][(3 + t) * DSMALL + n] =
        f32_bf16(q_small[((size_t)t * BQ + b) * DSMALL + n]);
  }
  if (tid < 16) lds_logit[tid] = 0.f;
  __syncthreads();

  // ---- phase B: lx = leaky(left@Wl^T+bl), rx1 = leaky(right@Wr^T+br)  (K=640) ----
  v8f aLx[2], aRx[2];
#pragma unroll
  for (int nt = 0; nt < 2; ++nt) { aLx[nt] = zero8; aRx[nt] = zero8; }

  for (int ks = 0; ks < 20; ++ks) {
    const int kbase = ks * 32;
    const int seg = kbase >> 7;                   // which 128-wide concat slice
    const int lk = kbase & 127;
    const int lcol = (seg == 0) ? 0 : (seg + 1) * DSMALL;  // hvi | rel,qs,qr,qt
    Frag fL, fR;
    loadA_bf16(fL, &sh.p.small[m16][lcol + lk], lane);
    if (seg == 0) loadA_bf16(fR, &sh.p.small[m16][1 * DSMALL + lk], lane);  // hvj
    else fR = fL;
#pragma unroll
    for (int nt = 0; nt < 2; ++nt) {
      Frag fB;
      loadB(fB, Wl_bf, 5 * DSMALL, (wave * 2 + nt) * 16, kbase, lane);
      aLx[nt] = wmma_bf16(fL, fB, aLx[nt]);
      loadB(fB, Wr_bf, 5 * DSMALL, (wave * 2 + nt) * 16, kbase, lane);
      aRx[nt] = wmma_bf16(fR, fB, aRx[nt]);
    }
  }
#pragma unroll
  for (int nt = 0; nt < 2; ++nt) {
    const int n = (wave * 2 + nt) * 16 + nlo;
    const float b_l = bl[n], b_r = br[n];
#pragma unroll
    for (int r = 0; r < 8; ++r) {
      const int m = m0 + r;
      sh.p.lx[m][n] = leaky(aLx[nt][r] + b_l);
      sh.p.rx1[m][n] = f32_bf16(leaky(aRx[nt][r] + b_r));
    }
  }
  __syncthreads();

  // ---- phase C: rx = rx1@Wc^T + bc ; logits[m] = sum_n lx*rx ----
  v8f aC[2];
#pragma unroll
  for (int nt = 0; nt < 2; ++nt) aC[nt] = zero8;
  for (int ks = 0; ks < 4; ++ks) {
    const int kbase = ks * 32;
    Frag fA;
    loadA_bf16(fA, &sh.p.rx1[m16][kbase], lane);
#pragma unroll
    for (int nt = 0; nt < 2; ++nt) {
      Frag fB;
      loadB(fB, Wc_bf, DSMALL, (wave * 2 + nt) * 16, kbase, lane);
      aC[nt] = wmma_bf16(fA, fB, aC[nt]);
    }
  }
#pragma unroll
  for (int nt = 0; nt < 2; ++nt) {
    const int n = (wave * 2 + nt) * 16 + nlo;
    const float b_c = bc[n];
#pragma unroll
    for (int r = 0; r < 8; ++r) {
      const int m = m0 + r;
      atomicAdd(&lds_logit[m], (aC[nt][r] + b_c) * sh.p.lx[m][n]);   // ds_add_f32
    }
  }
  __syncthreads();
  if (tid < 16) logits[e0 + tid] = lds_logit[tid];
}

// -------------------------------------------------- segment softmax pieces
__global__ void k_segmax(const float* __restrict__ logits,
                         const int* __restrict__ seg_src,
                         unsigned int* __restrict__ segmax) {
  int e = blockIdx.x * blockDim.x + threadIdx.x;
  if (e < E_EDGES) atomicMax(&segmax[seg_src[e]], fmap(logits[e]));
}

__global__ void k_exp(const float* __restrict__ logits, const int* __restrict__ seg_src,
                      const unsigned int* __restrict__ segmax,
                      float* __restrict__ exb, float* __restrict__ denom) {
  int e = blockIdx.x * blockDim.x + threadIdx.x;
  if (e < E_EDGES) {
    int s = seg_src[e];
    float ex = __expf(logits[e] - funmap(segmax[s]));
    exb[e] = ex;
    atomicAdd(&denom[s], ex);
  }
}

__global__ void k_prob_att(const float* __restrict__ exb, const float* __restrict__ denom,
                           const int* __restrict__ seg_src, const int* __restrict__ seg_tgt,
                           const float* __restrict__ node_attention,
                           float* __restrict__ prob, float* __restrict__ att_out) {
  int e = blockIdx.x * blockDim.x + threadIdx.x;
  if (e < E_EDGES) {
    float p = exb[e] / denom[seg_src[e]];
    prob[e] = p;
    atomicAdd(&att_out[seg_tgt[e]], p * node_attention[e]);
  }
}

// -------------------------------------------------- msg = hvj@Wlin^T, scatter
// one block = 16 edges x 512 outputs; 8 waves x 4 N-tiles; K = 512
__global__ __launch_bounds__(256) void k_msg(
    const float* __restrict__ memorized, const int* __restrict__ vj_idx,
    const int* __restrict__ seg_tgt, const unsigned short* __restrict__ Wlin_bf,
    const float* __restrict__ blin, const float* __restrict__ prob,
    float* __restrict__ upd_out) {
  __shared__ unsigned short stage[16][DD];   // 16 KB: hvj rows (bf16)

  const int e0 = blockIdx.x * 16;
  const int tid = threadIdx.x;
  const int lane = tid & 31;
  const int wave = tid >> 5;               // 0..7
  const int m16 = lane & 15;
  const int m0 = ((lane >> 4) & 1) * 8;
  const int nlo = lane & 15;
  const v8f zero8 = {0.f, 0.f, 0.f, 0.f, 0.f, 0.f, 0.f, 0.f};

  for (int i = tid; i < 16 * (DD / 2); i += 256) {
    const int e = i / (DD / 2);
    const int kp = i % (DD / 2);
    const float* src = memorized + (size_t)vj_idx[e0 + e] * DD;
    *(unsigned int*)&stage[e][2 * kp] = pack2_bf16(src[2 * kp], src[2 * kp + 1]);
  }
  __syncthreads();

  v8f acc[4];
#pragma unroll
  for (int nt = 0; nt < 4; ++nt) acc[nt] = zero8;

  for (int ks = 0; ks < 16; ++ks) {
    const int kbase = ks * 32;
    Frag fA;
    loadA_bf16(fA, &stage[m16][kbase], lane);
#pragma unroll
    for (int nt = 0; nt < 4; ++nt) {
      Frag fB;
      loadB(fB, Wlin_bf, DD, (wave * 4 + nt) * 16, kbase, lane);
      acc[nt] = wmma_bf16(fA, fB, acc[nt]);
    }
  }

  float pe[8];
  int tg[8];
#pragma unroll
  for (int r = 0; r < 8; ++r) {
    pe[r] = prob[e0 + m0 + r];
    tg[r] = seg_tgt[e0 + m0 + r];
  }
#pragma unroll
  for (int nt = 0; nt < 4; ++nt) {
    const int n = (wave * 4 + nt) * 16 + nlo;
    const float b = blin[n];
#pragma unroll
    for (int r = 0; r < 8; ++r)
      atomicAdd(&upd_out[(size_t)tg[r] * DD + n], pe[r] * (acc[nt][r] + b));
  }
}

// ---------------------------------------------------------------- launcher
extern "C" void kernel_launch(void* const* d_in, const int* in_sizes, int n_in,
                              void* d_out, int out_size, void* d_ws, size_t ws_size,
                              hipStream_t stream) {
  (void)in_sizes; (void)n_in; (void)ws_size;
  const float* node_attention = (const float*)d_in[0];
  const float* memorized = (const float*)d_in[1];
  const float* rel_emb = (const float*)d_in[2];
  const float* q_src = (const float*)d_in[3];
  const float* q_rel = (const float*)d_in[4];
  const float* q_time = (const float*)d_in[5];
  const int* eg_idx = (const int*)d_in[6];
  const int* vi_idx = (const int*)d_in[7];
  const int* vj_idx = (const int*)d_in[8];
  const int* seg_src = (const int*)d_in[9];
  const int* seg_tgt = (const int*)d_in[10];
  const float* Wproj = (const float*)d_in[11];
  const float* bproj = (const float*)d_in[12];
  const float* Wl = (const float*)d_in[13];
  const float* bl = (const float*)d_in[14];
  const float* Wr = (const float*)d_in[15];
  const float* br = (const float*)d_in[16];
  const float* Wc = (const float*)d_in[17];
  const float* bc = (const float*)d_in[18];
  const float* Wlin = (const float*)d_in[19];
  const float* blin = (const float*)d_in[20];

  char* ws = (char*)d_ws;
  unsigned short* Wproj_bf = (unsigned short*)(ws + 0);        // 128*512
  unsigned short* Wl_bf    = (unsigned short*)(ws + 131072);   // 128*640
  unsigned short* Wr_bf    = (unsigned short*)(ws + 294912);   // 128*640
  unsigned short* Wc_bf    = (unsigned short*)(ws + 458752);   // 128*128
  unsigned short* Wlin_bf  = (unsigned short*)(ws + 491520);   // 512*512
  float* q_small = (float*)(ws + 1015808);                     // 3*64*128
  float* logits  = (float*)(ws + 1114112);                     // E
  float* exb     = (float*)(ws + 1638400);                     // E
  float* prob    = (float*)(ws + 2162688);                     // E
  unsigned int* segmax = (unsigned int*)(ws + 2686976);        // N_SEG
  float* denom   = (float*)(ws + 2886976);                     // N_SEG

  float* att_out = (float*)d_out;
  float* upd_out = att_out + N_SEGS;

  // 1) weights -> bf16 (stay L2-resident)
  k_cvt_bf16<<<(DSMALL * DD + 255) / 256, 256, 0, stream>>>(Wproj, Wproj_bf, DSMALL * DD);
  k_cvt_bf16<<<(DSMALL * 5 * DSMALL + 255) / 256, 256, 0, stream>>>(Wl, Wl_bf, DSMALL * 5 * DSMALL);
  k_cvt_bf16<<<(DSMALL * 5 * DSMALL + 255) / 256, 256, 0, stream>>>(Wr, Wr_bf, DSMALL * 5 * DSMALL);
  k_cvt_bf16<<<(DSMALL * DSMALL + 255) / 256, 256, 0, stream>>>(Wc, Wc_bf, DSMALL * DSMALL);
  k_cvt_bf16<<<(DD * DD + 255) / 256, 256, 0, stream>>>(Wlin, Wlin_bf, DD * DD);
  // 2) tiny query projections (fp32)
  k_qproj<<<(3 * BQ * DSMALL + 255) / 256, 256, 0, stream>>>(q_src, q_rel, q_time, Wproj, bproj, q_small);
  // 3) init accumulators
  k_zero_f32<<<(out_size + 255) / 256, 256, 0, stream>>>(att_out, out_size);
  k_init_seg<<<(N_SEGS + 255) / 256, 256, 0, stream>>>(segmax, denom);
  // 4) fused per-edge logits (all-WMMA)
  k_logits<<<E_EDGES / 16, 128, 0, stream>>>(memorized, rel_emb, eg_idx, vi_idx, vj_idx,
                                             Wproj_bf, bproj, Wl_bf, bl, Wr_bf, br,
                                             Wc_bf, bc, q_small, logits);
  // 5) segment softmax + attention scatter
  k_segmax<<<(E_EDGES + 255) / 256, 256, 0, stream>>>(logits, seg_src, segmax);
  k_exp<<<(E_EDGES + 255) / 256, 256, 0, stream>>>(logits, seg_src, segmax, exb, denom);
  k_prob_att<<<(E_EDGES + 255) / 256, 256, 0, stream>>>(exb, denom, seg_src, seg_tgt,
                                                        node_attention, prob, att_out);
  // 6) dominant GEMM + weighted scatter into upd
  k_msg<<<E_EDGES / 16, 256, 0, stream>>>(memorized, vj_idx, seg_tgt, Wlin_bf, blin, prob, upd_out);
}
